// ContrastiveLoss_34248069218680
// MI455X (gfx1250) — matrix operand order repaired and checked by reading, not compiled, not run on hardware
//
#include <hip/hip_runtime.h>

// ---------------------------------------------------------------------------
// Fused contrastive loss for MI455X (gfx1250, wave32, WMMA):
//   sim = (z1/(||z1||*T)) . (z2/||z2||)^T, loss = -mean(logsumexp(sim, -1))
//
// Structure:
//  * split-bf16 (hi+lo) panels -> 3x V_WMMA_F32_16X16X32_BF16 per K=32 step
//    (~fp32 accuracy at bf16 matrix-core rate)
//  * grid = (64 row-blocks of M=128) x (8 j-chunks of 1024 cols):
//    all 8 waves of a block sweep identical B tiles (WGP$ reuse, ~1 GB total
//    L2 traffic), j-chunking keeps 4096 waves in flight.
//  * online logsumexp per wave; (max, sum) partials per (row, j-chunk) merged
//    by a second kernel; final kernel produces the scalar. No atomics.
//  * K-loop unroll capped at 2: bounds the software-pipeline live set to
//    < 256 VGPRs (avoids s_set_vgpr_msb overhead, keeps occupancy up).
// ---------------------------------------------------------------------------

typedef __bf16 v8bf  __attribute__((ext_vector_type(8)));
typedef __bf16 v16bf __attribute__((ext_vector_type(16)));
typedef float  v8f   __attribute__((ext_vector_type(8)));

union ABVec { v16bf v; v8bf h[2]; };

#define N_ROWS   8192
#define DIM      512
#define TEMP_INV 20.0f          // 1 / 0.05
#define NEG_BIG  (-3.0e38f)
#define JCHUNKS  8              // j-tiles split into 8 chunks of 64 tiles

#define WMMA_BF16(A, B, C)                                                   \
  __builtin_amdgcn_wmma_f32_16x16x32_bf16(false, (A), false, (B), (short)0,  \
                                          (C), false, false)

// ---------------------------------------------------------------------------
// Preprocess: per row, 1/||row|| (eps clamp irrelevant: ||x||~22.6 for
// N(0,1) 512-dim rows), fold `scale` (1/T for z1), split normalized fp32 into
// bf16 hi + bf16 lo residual panels. One wave per row.
// ---------------------------------------------------------------------------
__global__ __launch_bounds__(256) void prep_kernel(const float* __restrict__ z,
                                                   __bf16* __restrict__ hi,
                                                   __bf16* __restrict__ lo,
                                                   float scale) {
  const int lane = threadIdx.x & 31;
  const int row  = blockIdx.x * 8 + (threadIdx.x >> 5);

  const float* zr = z + (size_t)row * DIM;
  float x[16];
  float ss = 0.f;
#pragma unroll
  for (int i = 0; i < 16; ++i) {
    x[i] = zr[lane + 32 * i];
    ss += x[i] * x[i];
  }
#pragma unroll
  for (int off = 16; off > 0; off >>= 1) ss += __shfl_xor(ss, off, 32);

  const float inv = scale / sqrtf(ss);

  __bf16* hr = hi + (size_t)row * DIM;
  __bf16* lr = lo + (size_t)row * DIM;
#pragma unroll
  for (int i = 0; i < 16; ++i) {
    const float w  = x[i] * inv;
    const __bf16 h = (__bf16)w;                 // RNE
    hr[lane + 32 * i] = h;
    lr[lane + 32 * i] = (__bf16)(w - (float)h); // residual
  }
}

// ---------------------------------------------------------------------------
// Fused GEMM + online logsumexp over one (128-row, 64-j-tile) super-tile.
// Wave w owns rows mbase + 16w..+16; all waves iterate the same j-tiles
// (4 per pass, immediate-offset folded). 12 WMMAs : 20 b128 loads per K-step.
// A layout: lane l16 = row, K chunks at +8*half and +16+8*half.
// B layout: lane l16 = column (= z2 row, contiguous K), halves split K.
// ---------------------------------------------------------------------------
__global__ __launch_bounds__(256) void simlse_kernel(
    const __bf16* __restrict__ Ahi, const __bf16* __restrict__ Alo,
    const __bf16* __restrict__ Bhi, const __bf16* __restrict__ Blo,
    float* __restrict__ pmax, float* __restrict__ psum) {
  const int lane = threadIdx.x & 31;
  const int wave = threadIdx.x >> 5;
  const int half = lane >> 4;    // 0: lanes 0-15, 1: lanes 16-31
  const int l16  = lane & 15;

  const int mrow0 = blockIdx.x * 128 + wave * 16;  // this wave's 16 rows
  const int jt0   = blockIdx.y * 64;               // first j-tile of chunk

  const size_t arow = (size_t)(mrow0 + l16) * DIM + half * 8;

  float rmax[8], rsum[8];
#pragma unroll
  for (int r = 0; r < 8; ++r) { rmax[r] = NEG_BIG; rsum[r] = 0.f; }

  for (int jt = jt0; jt < jt0 + 64; jt += 4) {
    const size_t brow = (size_t)(jt * 16 + l16) * DIM + half * 8;

    v8f acc0 = {}, acc1 = {}, acc2 = {}, acc3 = {};
#pragma unroll 2
    for (int k = 0; k < DIM; k += 32) {
      ABVec ah, al;
      ah.h[0] = *(const v8bf*)(Ahi + arow + k);
      ah.h[1] = *(const v8bf*)(Ahi + arow + k + 16);
      al.h[0] = *(const v8bf*)(Alo + arow + k);
      al.h[1] = *(const v8bf*)(Alo + arow + k + 16);

      ABVec bh[4], bl[4];
#pragma unroll
      for (int t = 0; t < 4; ++t) {
        const size_t bo = brow + (size_t)t * 16 * DIM + k;
        bh[t].h[0] = *(const v8bf*)(Bhi + bo);
        bh[t].h[1] = *(const v8bf*)(Bhi + bo + 16);
        bl[t].h[0] = *(const v8bf*)(Blo + bo);
        bl[t].h[1] = *(const v8bf*)(Blo + bo + 16);
      }

      acc0 = WMMA_BF16(ah.v, bh[0].v, acc0);
      acc0 = WMMA_BF16(ah.v, bl[0].v, acc0);
      acc0 = WMMA_BF16(al.v, bh[0].v, acc0);
      acc1 = WMMA_BF16(ah.v, bh[1].v, acc1);
      acc1 = WMMA_BF16(ah.v, bl[1].v, acc1);
      acc1 = WMMA_BF16(al.v, bh[1].v, acc1);
      acc2 = WMMA_BF16(ah.v, bh[2].v, acc2);
      acc2 = WMMA_BF16(ah.v, bl[2].v, acc2);
      acc2 = WMMA_BF16(al.v, bh[2].v, acc2);
      acc3 = WMMA_BF16(ah.v, bh[3].v, acc3);
      acc3 = WMMA_BF16(ah.v, bl[3].v, acc3);
      acc3 = WMMA_BF16(al.v, bh[3].v, acc3);
    }

    // Online logsumexp update (1/T already folded into the A panel).
#pragma unroll
    for (int r = 0; r < 8; ++r) {
      const float s0 = acc0[r], s1 = acc1[r], s2 = acc2[r], s3 = acc3[r];
      const float nm =
          fmaxf(rmax[r], fmaxf(fmaxf(s0, s1), fmaxf(s2, s3)));
      rsum[r] = rsum[r] * __expf(rmax[r] - nm) + __expf(s0 - nm) +
                __expf(s1 - nm) + __expf(s2 - nm) + __expf(s3 - nm);
      rmax[r] = nm;
    }
  }

  // Merge the 16 column-lanes of each half-group (row-slot stays in-lane).
#pragma unroll
  for (int off = 8; off > 0; off >>= 1) {
#pragma unroll
    for (int r = 0; r < 8; ++r) {
      const float om = __shfl_xor(rmax[r], off, 32);
      const float os = __shfl_xor(rsum[r], off, 32);
      const float nm = fmaxf(rmax[r], om);
      rsum[r] = rsum[r] * __expf(rmax[r] - nm) + os * __expf(om - nm);
      rmax[r] = nm;
    }
  }

  // C/D layout: VGPR r -> row r (lanes 0-15) / row r+8 (lanes 16-31).
  if (l16 == 0) {
#pragma unroll
    for (int r = 0; r < 8; ++r) {
      const int row = mrow0 + half * 8 + r;
      pmax[(size_t)row * JCHUNKS + blockIdx.y] = rmax[r];
      psum[(size_t)row * JCHUNKS + blockIdx.y] = rsum[r];
    }
  }
}

// ---------------------------------------------------------------------------
// Merge the 8 j-chunk partials per row -> lse[row].
// ---------------------------------------------------------------------------
__global__ __launch_bounds__(256) void merge_kernel(const float* __restrict__ pmax,
                                                    const float* __restrict__ psum,
                                                    float* __restrict__ lse) {
  const int row = blockIdx.x * 256 + threadIdx.x;
  float m = pmax[(size_t)row * JCHUNKS];
  float s = psum[(size_t)row * JCHUNKS];
#pragma unroll
  for (int c = 1; c < JCHUNKS; ++c) {
    const float om = pmax[(size_t)row * JCHUNKS + c];
    const float os = psum[(size_t)row * JCHUNKS + c];
    const float nm = fmaxf(m, om);
    s = s * __expf(m - nm) + os * __expf(om - nm);
    m = nm;
  }
  lse[row] = m + logf(s);
}

// ---------------------------------------------------------------------------
// Final scalar: -mean(lse)
// ---------------------------------------------------------------------------
__global__ __launch_bounds__(256) void reduce_kernel(const float* __restrict__ lse,
                                                     float* __restrict__ out) {
  __shared__ float buf[256];
  float s = 0.f;
  for (int i = threadIdx.x; i < N_ROWS; i += 256) s += lse[i];
  buf[threadIdx.x] = s;
  __syncthreads();
  for (int off = 128; off > 0; off >>= 1) {
    if ((int)threadIdx.x < off) buf[threadIdx.x] += buf[threadIdx.x + off];
    __syncthreads();
  }
  if (threadIdx.x == 0) out[0] = -buf[0] / (float)N_ROWS;
}

extern "C" void kernel_launch(void* const* d_in, const int* in_sizes, int n_in,
                              void* d_out, int out_size, void* d_ws, size_t ws_size,
                              hipStream_t stream) {
  const float* z1 = (const float*)d_in[0];
  const float* z2 = (const float*)d_in[1];

  char* ws = (char*)d_ws;
  const size_t panel = (size_t)N_ROWS * DIM * sizeof(__bf16);     // 8 MB each
  __bf16* z1hi = (__bf16*)(ws + 0 * panel);
  __bf16* z1lo = (__bf16*)(ws + 1 * panel);
  __bf16* z2hi = (__bf16*)(ws + 2 * panel);
  __bf16* z2lo = (__bf16*)(ws + 3 * panel);
  char*   p    = ws + 4 * panel;
  float*  pmax = (float*)p;                                        // 256 KB
  float*  psum = (float*)(p + (size_t)N_ROWS * JCHUNKS * 4);       // 256 KB
  float*  lse  = (float*)(p + 2 * (size_t)N_ROWS * JCHUNKS * 4);   // 32 KB

  prep_kernel<<<N_ROWS / 8, 256, 0, stream>>>(z1, z1hi, z1lo, TEMP_INV);
  prep_kernel<<<N_ROWS / 8, 256, 0, stream>>>(z2, z2hi, z2lo, 1.0f);

  dim3 grid(N_ROWS / 128, JCHUNKS);
  simlse_kernel<<<grid, 256, 0, stream>>>(z1hi, z1lo, z2hi, z2lo, pmax, psum);

  merge_kernel<<<N_ROWS / 256, 256, 0, stream>>>(pmax, psum, lse);
  reduce_kernel<<<1, 256, 0, stream>>>(lse, (float*)d_out);
}
